// MegatronFMoEAdapter_21217138442731
// MI455X (gfx1250) — matrix operand order repaired and checked by reading, not compiled, not run on hardware
//
#include <hip/hip_runtime.h>
#include <math.h>

#define D_MODEL 768
#define H_FF    3072
#define NEXP    8

typedef __attribute__((ext_vector_type(16))) __bf16 v16bf;
typedef __attribute__((ext_vector_type(8)))  __bf16 v8bf;
typedef __attribute__((ext_vector_type(4)))  __bf16 v4bf;
typedef __attribute__((ext_vector_type(8)))  float  v8f;

static __device__ __forceinline__ v16bf cat16(v8bf lo, v8bf hi) {
    return __builtin_shufflevector(lo, hi, 0,1,2,3,4,5,6,7,8,9,10,11,12,13,14,15);
}

static __device__ __forceinline__ v8bf cvt8(float4 a, float4 b) {
    v8bf r;
    r[0] = (__bf16)a.x; r[1] = (__bf16)a.y; r[2] = (__bf16)a.z; r[3] = (__bf16)a.w;
    r[4] = (__bf16)b.x; r[5] = (__bf16)b.y; r[6] = (__bf16)b.z; r[7] = (__bf16)b.w;
    return r;
}

// ---------------------------------------------------------------------------
// Kernel 1: gate logits -> top-2 -> softmax -> per-expert token lists.
// One wave32 per token.
// ---------------------------------------------------------------------------
__global__ void gate_topk_kernel(const float* __restrict__ x,
                                 const float* __restrict__ gw,
                                 const float* __restrict__ gb,
                                 int* __restrict__ counts,
                                 int* __restrict__ perm,
                                 float* __restrict__ pwgt,
                                 int T) {
    const int wv   = threadIdx.x >> 5;
    const int lane = threadIdx.x & 31;
    const int token = blockIdx.x * 8 + wv;
    if (token >= T) return;

    const float* xr = x + (size_t)token * D_MODEL;
    float acc[NEXP];
#pragma unroll
    for (int e = 0; e < NEXP; ++e) acc[e] = 0.f;

    for (int k = lane; k < D_MODEL; k += 32) {
        float xv = xr[k];
#pragma unroll
        for (int e = 0; e < NEXP; ++e) acc[e] += xv * gw[e * D_MODEL + k];
    }
#pragma unroll
    for (int e = 0; e < NEXP; ++e) {
#pragma unroll
        for (int off = 16; off > 0; off >>= 1)
            acc[e] += __shfl_down(acc[e], off, 32);
    }
    if (lane == 0) {
        float lg[NEXP];
#pragma unroll
        for (int e = 0; e < NEXP; ++e) lg[e] = acc[e] + gb[e];
        int i0 = 0;
#pragma unroll
        for (int e = 1; e < NEXP; ++e) if (lg[e] > lg[i0]) i0 = e;
        int i1 = (i0 == 0) ? 1 : 0;
#pragma unroll
        for (int e = 0; e < NEXP; ++e) if (e != i0 && lg[e] > lg[i1]) i1 = e;
        // softmax over the two top logits (lg[i0] is the max)
        float e0 = expf(0.f);
        float e1 = expf(lg[i1] - lg[i0]);
        float inv = 1.f / (e0 + e1);
        float s0 = e0 * inv, s1 = e1 * inv;
        int p0 = atomicAdd(&counts[i0], 1);
        perm[i0 * T + p0] = token; pwgt[i0 * T + p0] = s0;
        int p1 = atomicAdd(&counts[i1], 1);
        perm[i1 * T + p1] = token; pwgt[i1 * T + p1] = s1;
    }
}

// ---------------------------------------------------------------------------
// Fused expert MLP: per (expert, 32-token tile), h = gelu(x@W1^T + b1) kept
// entirely in LDS (32x3072 bf16 = 192KB, fits CDNA5's 320KB/WGP), then
// y = h@W2^T + b2, scaled by the gate score, atomically added to out.
// ---------------------------------------------------------------------------
struct WRegs { float4 v[8]; };   // 64x128 fp32 tile / 256 threads = 32 floats each

static __device__ __forceinline__ WRegs gload_tile(const float* __restrict__ wsrc, int K,
                                                   int nbase, int kbase, int tid) {
    WRegs r;
    const int n  = tid >> 2;          // 0..63
    const int k0 = (tid & 3) << 5;    // 0,32,64,96
    const float4* s = (const float4*)(wsrc + (size_t)(nbase + n) * K + kbase + k0);
#pragma unroll
    for (int i = 0; i < 8; ++i) r.v[i] = s[i];
    return r;
}

static __device__ __forceinline__ void wstore_tile(const WRegs& r, __bf16* dst, int tid) {
    const int n  = tid >> 2;
    const int k0 = (tid & 3) << 5;
    v8bf* d = (v8bf*)(dst + n * 128 + k0);
#pragma unroll
    for (int i = 0; i < 4; ++i) d[i] = cvt8(r.v[2 * i], r.v[2 * i + 1]);
}

// 4 consecutive K=32 WMMA steps on one staged 64x128 bf16 weight buffer.
// A fragment per ISA 16-bit A(16x32) layout: lane<16 -> K {0..7,16..23},
// lane>=16 -> K {8..15,24..31} (two ds_load_b128 per step).
// B fragment per ISA 16-bit B(32x16) layout: lane%16 = N, lane<16 -> K {0..7,16..23},
// lane>=16 -> K {8..15,24..31} (two ds_load_b128 per step).
static __device__ __forceinline__ v8f kblock(const __bf16* __restrict__ alds, int astride,
                                             int kglobal, const __bf16* __restrict__ wbuf,
                                             int mrow, int base0, int nrow, int khalf, v8f acc) {
#pragma unroll
    for (int kk = 0; kk < 4; ++kk) {
        const __bf16* ap = alds + mrow * astride + kglobal + kk * 32 + base0;
        v8bf alo = *(const v8bf*)ap;
        v8bf ahi = *(const v8bf*)(ap + 16);
        const __bf16* bp = wbuf + nrow * 128 + kk * 32 + khalf * 8;
        v8bf blo = *(const v8bf*)bp;
        v8bf bhi = *(const v8bf*)(bp + 16);
        acc = __builtin_amdgcn_wmma_f32_16x16x32_bf16(false, cat16(alo, ahi),
                                                      false, cat16(blo, bhi),
                                                      (short)0, acc, false, false);
    }
    return acc;
}

__global__ void __launch_bounds__(256)
moe_expert_kernel(const float* __restrict__ x,
                  const float* __restrict__ w1,
                  const float* __restrict__ b1,
                  const float* __restrict__ w2,
                  const float* __restrict__ b2,
                  float* __restrict__ out,
                  const int* __restrict__ counts,
                  const int* __restrict__ perm,
                  const float* __restrict__ pwgt,
                  int T) {
    extern __shared__ char smem[];
    __bf16* xl  = (__bf16*)smem;            // 32 x 768  bf16  (48 KB)
    __bf16* hl  = xl + 32 * D_MODEL;        // 32 x 3072 bf16  (192 KB)
    __bf16* wl  = hl + 32 * H_FF;           // 2 x 64 x 128 bf16 (32 KB)
    int*    tok = (int*)(wl + 2 * 64 * 128);
    float*  twt = (float*)(tok + 32);

    const int e  = blockIdx.y;
    const int m0 = blockIdx.x * 32;
    const int cnt = counts[e];
    if (m0 >= cnt) return;                  // uniform early exit (no barriers yet)

    const int tid = threadIdx.x;
    if (tid < 32) {
        int slot = m0 + tid;
        if (slot < cnt) { tok[tid] = perm[e * T + slot]; twt[tid] = pwgt[e * T + slot]; }
        else            { tok[tid] = perm[e * T];        twt[tid] = 0.f; }  // pad: weight 0
    }
    __syncthreads();

    // Gather + fp32->bf16 convert the 32x768 token tile into LDS.
#pragma unroll
    for (int i = 0; i < 24; ++i) {
        int f4  = tid + i * 256;
        int row = f4 / 192;
        int c4  = f4 % 192;
        float4 v = ((const float4*)(x + (size_t)tok[row] * D_MODEL))[c4];
        v4bf h;
        h[0] = (__bf16)v.x; h[1] = (__bf16)v.y; h[2] = (__bf16)v.z; h[3] = (__bf16)v.w;
        *(v4bf*)(xl + row * D_MODEL + c4 * 4) = h;
    }
    __syncthreads();

    const int lane  = tid & 31;
    const int wv    = tid >> 5;
    const int mt    = wv >> 2;                  // 0..1: which 16-row half
    const int nt    = wv & 3;                   // 0..3: which 16-col slice
    const int mrow  = mt * 16 + (lane & 15);
    const int base0 = (lane & 16) ? 8 : 0;
    const int nrow  = nt * 16 + (lane & 15);
    const int khalf = lane >> 4;

    const float* w1e = w1 + (size_t)e * H_FF * D_MODEL;
    const float* w2e = w2 + (size_t)e * D_MODEL * H_FF;

    // ---------------- GEMM1: h = gelu(x @ W1^T + b1) ----------------
    for (int nb = 0; nb < H_FF / 64; ++nb) {
        v8f acc = {0.f, 0.f, 0.f, 0.f, 0.f, 0.f, 0.f, 0.f};
        WRegs r = gload_tile(w1e, D_MODEL, nb * 64, 0, tid);
        wstore_tile(r, wl, tid);
        __syncthreads();
        for (int s = 0; s < 6; ++s) {
            if (s + 1 < 6) r = gload_tile(w1e, D_MODEL, nb * 64, (s + 1) * 128, tid);
            acc = kblock(xl, D_MODEL, s * 128, wl + (s & 1) * (64 * 128),
                         mrow, base0, nrow, khalf, acc);
            if (s + 1 < 6) wstore_tile(r, wl + ((s + 1) & 1) * (64 * 128), tid);
            __syncthreads();
        }
        // epilogue: bias + exact-erf GELU, store bf16 into LDS h
        const int jg = nb * 64 + nt * 16 + (lane & 15);
        const float bb = b1[e * H_FF + jg];
#pragma unroll
        for (int rr = 0; rr < 8; ++rr) {
            float v = acc[rr] + bb;
            v = 0.5f * v * (1.f + erff(v * 0.70710678118f));
            int ml = ((lane >> 4) << 3) + rr;       // C layout: m = (lane/16)*8 + vgpr
            hl[(mt * 16 + ml) * H_FF + jg] = (__bf16)v;
        }
    }
    __syncthreads();

    // ---------------- GEMM2: y = h @ W2^T + b2, scaled combine ----------------
    for (int nb = 0; nb < D_MODEL / 64; ++nb) {
        v8f acc = {0.f, 0.f, 0.f, 0.f, 0.f, 0.f, 0.f, 0.f};
        WRegs r = gload_tile(w2e, H_FF, nb * 64, 0, tid);
        wstore_tile(r, wl, tid);
        __syncthreads();
        for (int s = 0; s < 24; ++s) {
            if (s + 1 < 24) r = gload_tile(w2e, H_FF, nb * 64, (s + 1) * 128, tid);
            acc = kblock(hl, H_FF, s * 128, wl + (s & 1) * (64 * 128),
                         mrow, base0, nrow, khalf, acc);
            if (s + 1 < 24) wstore_tile(r, wl + ((s + 1) & 1) * (64 * 128), tid);
            __syncthreads();
        }
        const int dg = nb * 64 + nt * 16 + (lane & 15);
        const float bb = b2[e * D_MODEL + dg];
#pragma unroll
        for (int rr = 0; rr < 8; ++rr) {
            int ml = ((lane >> 4) << 3) + rr;
            int mlocal = mt * 16 + ml;
            float v = (acc[rr] + bb) * twt[mlocal];
            atomicAdd(out + (size_t)tok[mlocal] * D_MODEL + dg, v);
        }
    }
}

// ---------------------------------------------------------------------------
extern "C" void kernel_launch(void* const* d_in, const int* in_sizes, int n_in,
                              void* d_out, int out_size, void* d_ws, size_t ws_size,
                              hipStream_t stream) {
    (void)n_in; (void)out_size; (void)ws_size;
    const float* x    = (const float*)d_in[0];
    const float* gw   = (const float*)d_in[1];
    const float* gb   = (const float*)d_in[2];
    const float* w1   = (const float*)d_in[3];
    const float* b1   = (const float*)d_in[4];
    const float* w2   = (const float*)d_in[5];
    const float* b2   = (const float*)d_in[6];
    const float* bias = (const float*)d_in[7];
    float* out = (float*)d_out;

    const int T = in_sizes[0] / D_MODEL;    // 4096

    // workspace: [counts: 8 ints (padded 128B)][perm: E*T ints][pwgt: E*T floats]
    int*   counts = (int*)d_ws;
    int*   perm   = (int*)((char*)d_ws + 128);
    float* pwgt   = (float*)((char*)d_ws + 128 + (size_t)NEXP * T * sizeof(int));

    hipMemsetAsync(d_out, 0, (size_t)T * D_MODEL * sizeof(float), stream);
    hipMemsetAsync(counts, 0, 128, stream);

    gate_topk_kernel<<<T / 8, 256, 0, stream>>>(x, gw, gb, counts, perm, pwgt, T);

    const size_t smem = (size_t)(32 * D_MODEL + 32 * H_FF + 2 * 64 * 128) * 2
                      + 32 * sizeof(int) + 32 * sizeof(float);   // ~279 KB < 320 KB/WGP
    dim3 grid((T + 31) / 32, NEXP);
    moe_expert_kernel<<<grid, 256, smem, stream>>>(x, w1, b1, w2, b2, out,
                                                   counts, perm, pwgt, T);

    // tuple output: append `bias` after the [T, D] tensor
    hipMemcpyAsync(out + (size_t)T * D_MODEL, bias, D_MODEL * sizeof(float),
                   hipMemcpyDeviceToDevice, stream);
}
